// AttenLayer_2233382993950
// MI455X (gfx1250) — matrix operand order repaired and checked by reading, not compiled, not run on hardware
//
#include <hip/hip_runtime.h>

#define B_  64
#define S_  2048
#define H_  512
#define A_  512

typedef __attribute__((ext_vector_type(16))) __bf16 v16bf;
typedef __attribute__((ext_vector_type(8)))  float  v8f;

#if __has_builtin(__builtin_amdgcn_tanhf)
#define TANHF(x) __builtin_amdgcn_tanhf(x)     // native v_tanh_f32 (TRANS32)
#else
#define TANHF(x) tanhf(x)
#endif

static __device__ __forceinline__ unsigned short f2bf(float x) {
    unsigned int u = __float_as_uint(x);
    // round-to-nearest-even fp32 -> bf16
    unsigned int r = (u + 0x7FFFu + ((u >> 16) & 1u)) >> 16;
    return (unsigned short)r;
}

// ---------------------------------------------------------------------------
// Kernel 1: convert proj_w (A_ x H_, fp32 row-major) to bf16 in workspace.
// ---------------------------------------------------------------------------
__global__ void convert_w_kernel(const float* __restrict__ w,
                                 unsigned short* __restrict__ wbf) {
    int i = (blockIdx.x * 256 + threadIdx.x) * 4;       // 4 elems per thread
    float4 f = *(const float4*)(w + i);
    unsigned int lo = (unsigned int)f2bf(f.x) | ((unsigned int)f2bf(f.y) << 16);
    unsigned int hi = (unsigned int)f2bf(f.z) | ((unsigned int)f2bf(f.w) << 16);
    uint2 p; p.x = lo; p.y = hi;
    *(uint2*)(wbf + i) = p;
}

// ---------------------------------------------------------------------------
// Kernel 2: fused GEMM + tanh + context-dot -> scores[b,s].
// One wave handles 16 rows of S (an M-tile). N = A_ = 512 processed as 8
// passes of 4 N-tiles (4 accumulators) so each A fragment (LDS) feeds 4
// v_wmma_f32_16x16x32_bf16 ops. K = H_ = 512 (16 WMMA K-steps of 32).
// ---------------------------------------------------------------------------
#define WAVES_PER_WG 8
#define LDS_STRIDE   520   // 512 bf16 + 8 pad (16B) -> conflict-friendly rows
#define NBLK 4             // N tiles processed together

__global__ void scores_kernel(const float* __restrict__ nn,
                              const unsigned short* __restrict__ wbf,
                              const float* __restrict__ ctx,
                              const float* __restrict__ bias,
                              float* __restrict__ scores) {
    extern __shared__ unsigned short smem[];

    const int lane   = threadIdx.x & 31;
    const int waveId = threadIdx.x >> 5;
    const int mtile  = blockIdx.x * WAVES_PER_WG + waveId;   // 0 .. B*S/16-1
    const int b      = mtile >> 7;                           // / (S_/16)
    const int srow0  = (mtile & 127) << 4;

    unsigned short* atile = smem + waveId * 16 * LDS_STRIDE;

    // ---- Stage this wave's 16 x 512 fp32 tile into LDS as bf16 ----
    const float* src = nn + ((size_t)b * S_ + srow0) * H_;
    #pragma unroll 4
    for (int it = 0; it < 64; ++it) {
        int idx = it * 128 + lane * 4;          // 0..8191, 4 floats each
        int r = idx >> 9;                       // / 512
        int c = idx & 511;
        float4 f = *(const float4*)(src + (size_t)r * H_ + c);
        unsigned int lo = (unsigned int)f2bf(f.x) | ((unsigned int)f2bf(f.y) << 16);
        unsigned int hi = (unsigned int)f2bf(f.z) | ((unsigned int)f2bf(f.w) << 16);
        uint2 p; p.x = lo; p.y = hi;
        *(uint2*)(atile + r * LDS_STRIDE + c) = p;
    }
    __syncthreads();   // uniform across the WG; also flushes DScnt

    const int row   = lane & 15;    // A row m / B,C column n this lane serves
    const int khalf = lane >> 4;    // which K half-group this lane holds

    float part[8];
    #pragma unroll
    for (int v = 0; v < 8; ++v) part[v] = 0.0f;

    for (int nt = 0; nt < 32; nt += NBLK) {
        // W rows for the NBLK tiles: a = (nt+j)*16 + row
        const unsigned short* wrow0 = wbf + (size_t)(nt * 16 + row) * H_;

        v8f acc[NBLK];
        #pragma unroll
        for (int j = 0; j < NBLK; ++j)
            acc[j] = (v8f){0.f, 0.f, 0.f, 0.f, 0.f, 0.f, 0.f, 0.f};

        #pragma unroll 2
        for (int k0 = 0; k0 < H_; k0 += 32) {
            // A fragment (16x32 bf16): lane holds row, 16 contiguous K at
            // k0 + khalf*16                       (matches B order)
            union { v16bf v; uint4 q[2]; } af;
            const unsigned short* ap = atile + row * LDS_STRIDE + k0 + khalf * 16;
            af.q[0] = *(const uint4*)ap;          // ds_load_b128
            af.q[1] = *(const uint4*)(ap + 8);    // ds_load_b128

            // B fragments (32x16 bf16): lane holds col a, 16 contiguous K at
            // k0 + khalf*16                       (same order as A)
            #pragma unroll
            for (int j = 0; j < NBLK; ++j) {
                union { v16bf v; uint4 q[2]; } bfr;
                const unsigned short* bp =
                    wrow0 + (size_t)j * 16 * H_ + k0 + khalf * 16;
                bfr.q[0] = *(const uint4*)bp;     // global_load_b128
                bfr.q[1] = *(const uint4*)(bp + 8);
                acc[j] = __builtin_amdgcn_wmma_f32_16x16x32_bf16(
                             false, af.v, false, bfr.v, (short)0, acc[j],
                             false, false);
            }
        }

        // Fuse: bias + tanh + context weighting, accumulate per-row partials.
        // C layout: acc[j][v] is element (m = v + 8*khalf, n = row),
        //           a = (nt+j)*16 + row.
        #pragma unroll
        for (int j = 0; j < NBLK; ++j) {
            const int acol = (nt + j) * 16 + row;
            const float ba = bias[acol];
            const float ca = ctx[acol];
            #pragma unroll
            for (int v = 0; v < 8; ++v)
                part[v] += ca * TANHF(acc[j][v] + ba);
        }
    }

    // Reduce over the 16 lanes of each half (sum over n) -> 16 scores / wave.
    #pragma unroll
    for (int m = 1; m < 16; m <<= 1) {
        #pragma unroll
        for (int v = 0; v < 8; ++v)
            part[v] += __shfl_xor(part[v], m, 32);
    }
    if (row == 0) {
        const int mbase = khalf * 8;
        #pragma unroll
        for (int v = 0; v < 8; ++v)
            scores[(size_t)b * S_ + srow0 + mbase + v] = part[v];
    }
}

// ---------------------------------------------------------------------------
// Kernel 3: length-masked softmax, one block per batch row.
// ---------------------------------------------------------------------------
__global__ void softmax_kernel(const float* __restrict__ scores,
                               float* __restrict__ atten,
                               const int* __restrict__ lens) {
    __shared__ float red[256];
    const int b   = blockIdx.x;
    const int tid = threadIdx.x;
    int len = lens[b];
    if (len < 1)  len = 1;
    if (len > S_) len = S_;

    const float* srow = scores + (size_t)b * S_;
    float* arow = atten + (size_t)b * S_;

    // max over valid range
    float m = -3.402823466e+38f;
    for (int s = tid; s < len; s += 256) m = fmaxf(m, srow[s]);
    red[tid] = m;
    __syncthreads();
    for (int off = 128; off > 0; off >>= 1) {
        if (tid < off) red[tid] = fmaxf(red[tid], red[tid + off]);
        __syncthreads();
    }
    const float mx = red[0];
    __syncthreads();

    // exp + sum; padded positions written as exact zero
    float sum = 0.0f;
    for (int s = tid; s < S_; s += 256) {
        float e = 0.0f;
        if (s < len) { e = __expf(srow[s] - mx); sum += e; }
        arow[s] = e;
    }
    red[tid] = sum;
    __syncthreads();
    for (int off = 128; off > 0; off >>= 1) {
        if (tid < off) red[tid] += red[tid + off];
        __syncthreads();
    }
    const float inv = 1.0f / red[0];

    for (int s = tid; s < len; s += 256) arow[s] *= inv;
}

// ---------------------------------------------------------------------------
// Kernel 4: out partials: part[b][sp][h] = sum over a 256-step S slice.
// grid = (B_, H_/256, S_/256)
// ---------------------------------------------------------------------------
__global__ void wsum_partial_kernel(const float* __restrict__ nn,
                                    const float* __restrict__ atten,
                                    float* __restrict__ part) {
    const int b  = blockIdx.x;
    const int h  = blockIdx.y * 256 + threadIdx.x;
    const int sp = blockIdx.z;
    const int s0 = sp * 256;

    const float* base = nn + ((size_t)b * S_ + s0) * H_ + h;
    const float* at   = atten + (size_t)b * S_ + s0;

    float acc = 0.0f;
    #pragma unroll 4
    for (int s = 0; s < 256; ++s)
        acc = fmaf(at[s], base[(size_t)s * H_], acc);

    part[((size_t)b * 8 + sp) * H_ + h] = acc;
}

// ---------------------------------------------------------------------------
// Kernel 5: deterministic reduction of the 8 S-slices -> out[b,h].
// ---------------------------------------------------------------------------
__global__ void wsum_reduce_kernel(const float* __restrict__ part,
                                   float* __restrict__ out) {
    const int i = blockIdx.x * 256 + threadIdx.x;   // 0 .. B_*H_-1
    const int b = i >> 9;
    const int h = i & (H_ - 1);
    float acc = 0.0f;
    #pragma unroll
    for (int sp = 0; sp < 8; ++sp)
        acc += part[((size_t)b * 8 + sp) * H_ + h];
    out[i] = acc;
}

// ---------------------------------------------------------------------------
extern "C" void kernel_launch(void* const* d_in, const int* in_sizes, int n_in,
                              void* d_out, int out_size, void* d_ws, size_t ws_size,
                              hipStream_t stream) {
    const float* nn   = (const float*)d_in[0];   // (B,S,H) fp32
    const int*   lens = (const int*)d_in[1];     // (B,)
    const float* ctx  = (const float*)d_in[2];   // (A,)
    const float* pw   = (const float*)d_in[3];   // (A,H)
    const float* pb   = (const float*)d_in[4];   // (A,)
    float* out = (float*)d_out;                  // (B,H)

    char* ws = (char*)d_ws;
    unsigned short* wbf = (unsigned short*)ws;                        // 512 KB
    float* scores = (float*)(ws + (size_t)A_ * H_ * 2);               // 512 KB
    float* atten  = scores + (size_t)B_ * S_;                         // 512 KB
    float* part   = atten  + (size_t)B_ * S_;                         // 1 MB

    // 1) W -> bf16
    convert_w_kernel<<<(A_ * H_) / (256 * 4), 256, 0, stream>>>(pw, wbf);

    // 2) fused GEMM(tanh) + context dot -> scores
    const int mtiles = B_ * (S_ / 16);                     // 8192 waves
    const size_t lds_bytes = (size_t)WAVES_PER_WG * 16 * LDS_STRIDE * 2; // 133120
    scores_kernel<<<mtiles / WAVES_PER_WG, WAVES_PER_WG * 32, lds_bytes, stream>>>(
        nn, wbf, ctx, pb, scores);

    // 3) masked softmax
    softmax_kernel<<<B_, 256, 0, stream>>>(scores, atten, lens);

    // 4) weighted sum partials over S slices
    dim3 g4(B_, H_ / 256, S_ / 256);
    wsum_partial_kernel<<<g4, 256, 0, stream>>>(nn, atten, part);

    // 5) reduce partials -> out
    wsum_reduce_kernel<<<(B_ * H_) / 256, 256, 0, stream>>>(part, out);
}
